// TopoFlowAttention_47184510714614
// MI455X (gfx1250) — compile-verified, matrix operand order checked by reading, not run on hardware
//
#include <hip/hip_runtime.h>
#include <hip/hip_bf16.h>
#include <math.h>

// ---------------------------------------------------------------------------
// Types for CDNA5 WMMA (wave32, 16x16x32 bf16 -> f32)
// ---------------------------------------------------------------------------
typedef __attribute__((ext_vector_type(16))) __bf16 v16bf;
typedef __attribute__((ext_vector_type(8)))  __bf16 v8bf;
typedef __attribute__((ext_vector_type(4)))  __bf16 v4bf;
typedef __attribute__((ext_vector_type(8)))  float  v8f;
typedef __attribute__((ext_vector_type(4)))  float  f4;

#define NUM_HEADS 12
#define HEAD_DIM  64
#define DIM       768
#define SEQ_N     1024
#define BATCH     4

__device__ inline v8f wmma_bf16(v16bf a, v16bf b, v8f c) {
  // D = A(16x32 bf16) * B(32x16 bf16) + C(16x16 f32)
  return __builtin_amdgcn_wmma_f32_16x16x32_bf16(
      /*neg_a=*/false, a, /*neg_b=*/false, b,
      /*c_mod=*/(short)0, c, /*reuse_a=*/false, /*reuse_b=*/false);
}

// Load a 16x32 bf16 A-fragment (or the mirrored B-fragment) from a row-major
// tile.  Per the CDNA5 ISA 16-bit matrix layout:
//   lanes 0-15 : row = lane,    K = {0..7, 16..23}
//   lanes 16-31: row = lane-16, K = {8..15, 24..31}
// Each lane does two contiguous 16-byte loads.
__device__ inline v16bf load_frag(const __bf16* row0, int ld, int koff, int lane) {
  int r  = lane & 15;
  int k0 = koff + ((lane & 16) ? 8 : 0);
  const __bf16* p = row0 + (size_t)r * ld + k0;
  v8bf lo = *reinterpret_cast<const v8bf*>(p);
  v8bf hi = *reinterpret_cast<const v8bf*>(p + 16);
  v16bf f;
#pragma unroll
  for (int i = 0; i < 8; ++i) { f[i] = lo[i]; f[i + 8] = hi[i]; }
  return f;
}

__device__ inline v4bf cvt4(f4 v) {
  v4bf t;
  t[0] = (__bf16)v[0]; t[1] = (__bf16)v[1];
  t[2] = (__bf16)v[2]; t[3] = (__bf16)v[3];
  return t;
}

// ---------------------------------------------------------------------------
// Kernel 1: relative-position bucket index table  idx[b][i][j] in [0,1024)
// ---------------------------------------------------------------------------
__device__ inline int bucket_pos(int rel) {
  int n   = -rel;
  int ret = (n < 0) ? 16 : 0;     // NUM_BUCKETS/2
  n = (n < 0) ? -n : n;
  int v;
  if (n < 8) {                    // max_exact
    v = n;
  } else {
    float nf = (float)n;
    // log(n/8)/log(128/8) * 8, truncated
    v = 8 + (int)(__logf(nf * 0.125f) * (8.0f / __logf(16.0f)));
    if (v > 15) v = 15;
  }
  return ret + v;
}

__global__ __launch_bounds__(256) void bucket_idx_kernel(
    const float* __restrict__ coords, unsigned short* __restrict__ idx) {
  size_t t = (size_t)blockIdx.x * blockDim.x + threadIdx.x;
  const size_t total = (size_t)BATCH * SEQ_N * SEQ_N;
  if (t >= total) return;
  int j = (int)(t % SEQ_N);
  size_t r = t / SEQ_N;
  int i = (int)(r % SEQ_N);
  int b = (int)(r / SEQ_N);
  const float* ci = coords + ((size_t)b * SEQ_N + i) * 2;
  const float* cj = coords + ((size_t)b * SEQ_N + j) * 2;
  int xi = (int)(ci[0] * 128.0f), yi = (int)(ci[1] * 128.0f);
  int xj = (int)(cj[0] * 128.0f), yj = (int)(cj[1] * 128.0f);
  int bx = bucket_pos(xi - xj);
  int by = bucket_pos(yi - yj);
  idx[t] = (unsigned short)(bx * 32 + by);
}

// ---------------------------------------------------------------------------
// Kernel 2: QKV GEMM  [4096 x 768] * [768 x 2304] -> q/k/vT (bf16, scattered)
//   q  : [B,H,N,hd]   (scaled by hd^-0.5)
//   k  : [B,H,N,hd]
//   vT : [B,H,hd,N]   (pre-transposed for the P*V B-fragment path)
// Block: 128 threads (4 waves). Tile: 64(M) x 64(N) x 32(K).
// Staging: float4 global loads -> packed bf16 -> ds_store_b64.
// ---------------------------------------------------------------------------
__global__ __launch_bounds__(128) void qkv_gemm_kernel(
    const float* __restrict__ x, const float* __restrict__ qkv_w,
    const float* __restrict__ qkv_b,
    __bf16* __restrict__ qbuf, __bf16* __restrict__ kbuf,
    __bf16* __restrict__ vtbuf) {
  __shared__ __bf16 sA[64 * 32];   // [m][k]
  __shared__ __bf16 sB[64 * 32];   // [n][k]  (B^T rows == qkv_w rows)
  int tid  = threadIdx.x;
  int lane = tid & 31;
  int wave = tid >> 5;
  int mBase = blockIdx.y * 64;
  int nBase = blockIdx.x * 64;

  // per-thread staging coordinates: 4 float4 chunks each for A and B
  int rr[4], cc[4];
#pragma unroll
  for (int i = 0; i < 4; ++i) {
    int lin4 = tid + i * 128;         // float4 index 0..511
    rr[i] = lin4 >> 3;                // 8 float4 per 32-float row
    cc[i] = (lin4 & 7) << 2;
  }

  v8f acc[4] = {{}, {}, {}, {}};

  for (int kb = 0; kb < DIM; kb += 32) {
    // ---- bulk load (global_load_b128), then convert+store (ds_store_b64)
    f4 va[4], vb[4];
#pragma unroll
    for (int i = 0; i < 4; ++i) {
      va[i] = *(const f4*)&x[(size_t)(mBase + rr[i]) * DIM + kb + cc[i]];
      vb[i] = *(const f4*)&qkv_w[(size_t)(nBase + rr[i]) * DIM + kb + cc[i]];
    }
    if (kb + 32 < DIM) {   // prefetch next k-tile (global_prefetch_b8)
      __builtin_prefetch(&x[(size_t)(mBase + rr[0]) * DIM + kb + 32 + cc[0]], 0, 0);
      __builtin_prefetch(&qkv_w[(size_t)(nBase + rr[0]) * DIM + kb + 32 + cc[0]], 0, 0);
    }
#pragma unroll
    for (int i = 0; i < 4; ++i) {
      *(v4bf*)&sA[rr[i] * 32 + cc[i]] = cvt4(va[i]);
      *(v4bf*)&sB[rr[i] * 32 + cc[i]] = cvt4(vb[i]);
    }
    __syncthreads();

    v16bf a = load_frag(sA + wave * 16 * 32, 32, 0, lane);
#pragma unroll
    for (int nt = 0; nt < 4; ++nt) {
      v16bf bf = load_frag(sB + nt * 16 * 32, 32, 0, lane);
      acc[nt] = wmma_bf16(a, bf, acc[nt]);
    }
    __syncthreads();
  }

  // Epilogue: scatter into q / k / vT
  int hi = (lane & 16) ? 8 : 0;
  int nl = lane & 15;
#pragma unroll
  for (int nt = 0; nt < 4; ++nt) {
#pragma unroll
    for (int v = 0; v < 8; ++v) {
      int grow = mBase + wave * 16 + v + hi;     // 0..4095 token row
      int gc   = nBase + nt * 16 + nl;           // 0..2303
      float val = acc[nt][v] + qkv_b[gc];
      int part = (gc >= 2 * DIM) ? 2 : ((gc >= DIM) ? 1 : 0);
      int rem  = gc - part * DIM;
      int h    = rem >> 6;
      int d    = rem & 63;
      int b    = grow >> 10;
      int ntok = grow & 1023;
      size_t bh = (size_t)(b * NUM_HEADS + h);
      if (part == 0) {
        qbuf[(bh * SEQ_N + ntok) * HEAD_DIM + d] = (__bf16)(val * 0.125f);
      } else if (part == 1) {
        kbuf[(bh * SEQ_N + ntok) * HEAD_DIM + d] = (__bf16)val;
      } else {
        vtbuf[(bh * HEAD_DIM + d) * SEQ_N + ntok] = (__bf16)val;
      }
    }
  }
}

// ---------------------------------------------------------------------------
// Kernel 3: flash attention.  One wave = 16 query rows, block = 4 waves,
// grid = B*H*16.  Streams keys in blocks of 32; online softmax; biases fused.
// ---------------------------------------------------------------------------
__global__ __launch_bounds__(128) void attn_kernel(
    const __bf16* __restrict__ q, const __bf16* __restrict__ k,
    const __bf16* __restrict__ vt, const unsigned short* __restrict__ idx,
    const float* __restrict__ bias_table, const float* __restrict__ elev,
    const float* __restrict__ alpha, __bf16* __restrict__ aout) {
  __shared__ __bf16 sP[4][16 * 32];   // per-wave P staging (C-layout -> A-layout)
  int lane = threadIdx.x & 31;
  int wave = threadIdx.x >> 5;
  int bh = blockIdx.x >> 4;           // b*12 + h
  int qt = blockIdx.x & 15;
  int b = bh / NUM_HEADS;
  int h = bh - b * NUM_HEADS;
  int qtok = qt * 64 + wave * 16;     // first query row of this wave

  const __bf16* qp = q + ((size_t)bh * SEQ_N + qtok) * HEAD_DIM;
  v16bf qa0 = load_frag(qp, HEAD_DIM, 0, lane);
  v16bf qa1 = load_frag(qp, HEAD_DIM, 32, lane);

  int hi = (lane & 16) ? 8 : 0;
  int nl = lane & 15;
  float negalpha = -alpha[0];

  float eq[8];
#pragma unroll
  for (int v = 0; v < 8; ++v) eq[v] = elev[b * SEQ_N + qtok + v + hi];

  float mrun[8], lrun[8];
#pragma unroll
  for (int v = 0; v < 8; ++v) { mrun[v] = -3.0e38f; lrun[v] = 0.0f; }
  v8f o0 = {}, o1 = {}, o2 = {}, o3 = {};

  const __bf16* kbase = k + (size_t)bh * SEQ_N * HEAD_DIM;
  const __bf16* vbase = vt + (size_t)bh * HEAD_DIM * SEQ_N;
  const unsigned short* idxb = idx + (size_t)b * SEQ_N * SEQ_N;

  for (int kb = 0; kb < SEQ_N; kb += 32) {
    // ---- S = Q * K^T   (two 16x16 score tiles, keys kb..+15 and kb+16..+31)
    v8f s0 = {}, s1 = {};
    {
      v16bf bf = load_frag(kbase + (size_t)kb * HEAD_DIM, HEAD_DIM, 0, lane);
      s0 = wmma_bf16(qa0, bf, s0);
      bf = load_frag(kbase + (size_t)kb * HEAD_DIM, HEAD_DIM, 32, lane);
      s0 = wmma_bf16(qa1, bf, s0);
      bf = load_frag(kbase + (size_t)(kb + 16) * HEAD_DIM, HEAD_DIM, 0, lane);
      s1 = wmma_bf16(qa0, bf, s1);
      bf = load_frag(kbase + (size_t)(kb + 16) * HEAD_DIM, HEAD_DIM, 32, lane);
      s1 = wmma_bf16(qa1, bf, s1);
    }

    // ---- fused biases (relative-position table gather + elevation)
    int j0 = kb + nl, j1 = kb + 16 + nl;
    float ej0 = elev[b * SEQ_N + j0];
    float ej1 = elev[b * SEQ_N + j1];
#pragma unroll
    for (int v = 0; v < 8; ++v) {
      int qrow = qtok + v + hi;
      const unsigned short* ir = idxb + (size_t)qrow * SEQ_N;
      float bias0 = bias_table[(int)ir[j0] * NUM_HEADS + h];
      float bias1 = bias_table[(int)ir[j1] * NUM_HEADS + h];
      float ed0 = (ej0 - eq[v]) * 1.0e-3f;
      float ed1 = (ej1 - eq[v]) * 1.0e-3f;
      float eb0 = fmaxf(negalpha * fmaxf(ed0, 0.0f), -10.0f);
      float eb1 = fmaxf(negalpha * fmaxf(ed1, 0.0f), -10.0f);
      s0[v] += bias0 + eb0;
      s1[v] += bias1 + eb1;
    }

    // ---- online softmax (row reductions across the 16-lane half-wave)
    float corr[8];
#pragma unroll
    for (int v = 0; v < 8; ++v) {
      float mt = fmaxf(s0[v], s1[v]);
#pragma unroll
      for (int off = 8; off > 0; off >>= 1)
        mt = fmaxf(mt, __shfl_xor(mt, off, 32));
      float nm = fmaxf(mrun[v], mt);
      corr[v] = __expf(mrun[v] - nm);
      mrun[v] = nm;
      float p0 = __expf(s0[v] - nm);
      float p1 = __expf(s1[v] - nm);
      s0[v] = p0; s1[v] = p1;
      float rs = p0 + p1;
#pragma unroll
      for (int off = 8; off > 0; off >>= 1)
        rs += __shfl_xor(rs, off, 32);
      lrun[v] = lrun[v] * corr[v] + rs;
    }
#pragma unroll
    for (int v = 0; v < 8; ++v) {
      o0[v] *= corr[v]; o1[v] *= corr[v]; o2[v] *= corr[v]; o3[v] *= corr[v];
    }

    // ---- P (C-layout) -> LDS -> A-fragment
    __bf16* P = sP[wave];
#pragma unroll
    for (int v = 0; v < 8; ++v) {
      int m = v + hi;
      P[m * 32 + nl]      = (__bf16)s0[v];
      P[m * 32 + 16 + nl] = (__bf16)s1[v];
    }
    asm volatile("s_wait_dscnt 0" ::: "memory");   // wave-local store->load
    v16bf pa = load_frag(P, 32, 0, lane);

    // ---- O += P * V    (V^T rows are hd -> contiguous B-fragment loads)
    o0 = wmma_bf16(pa, load_frag(vbase + (size_t)0  * SEQ_N + kb, SEQ_N, 0, lane), o0);
    o1 = wmma_bf16(pa, load_frag(vbase + (size_t)16 * SEQ_N + kb, SEQ_N, 0, lane), o1);
    o2 = wmma_bf16(pa, load_frag(vbase + (size_t)32 * SEQ_N + kb, SEQ_N, 0, lane), o2);
    o3 = wmma_bf16(pa, load_frag(vbase + (size_t)48 * SEQ_N + kb, SEQ_N, 0, lane), o3);
  }

  // ---- normalize and store [B,N,H*hd] bf16 (row-major for the proj GEMM)
#pragma unroll
  for (int v = 0; v < 8; ++v) {
    float inv = 1.0f / lrun[v];
    int m = v + hi;
    size_t row = ((size_t)b * SEQ_N + qtok + m) * DIM + h * HEAD_DIM + nl;
    aout[row + 0]  = (__bf16)(o0[v] * inv);
    aout[row + 16] = (__bf16)(o1[v] * inv);
    aout[row + 32] = (__bf16)(o2[v] * inv);
    aout[row + 48] = (__bf16)(o3[v] * inv);
  }
}

// ---------------------------------------------------------------------------
// Kernel 4: output projection  [4096 x 768](bf16) * [768 x 768] + b -> f32
// A tile is already bf16 in memory -> copy to LDS with the CDNA5 async
// global->LDS path (ASYNCcnt, no VGPR round trip).
// ---------------------------------------------------------------------------
__global__ __launch_bounds__(128) void proj_gemm_kernel(
    const __bf16* __restrict__ aA, const float* __restrict__ proj_w,
    const float* __restrict__ proj_b, float* __restrict__ out) {
  __shared__ __bf16 sA[64 * 32];
  __shared__ __bf16 sB[64 * 32];
  int tid  = threadIdx.x;
  int lane = tid & 31;
  int wave = tid >> 5;
  int mBase = blockIdx.y * 64;
  int nBase = blockIdx.x * 64;

  int rr[4], cc[4];
#pragma unroll
  for (int i = 0; i < 4; ++i) {
    int lin4 = tid + i * 128;
    rr[i] = lin4 >> 3;
    cc[i] = (lin4 & 7) << 2;
  }
  // async A-copy coordinates: 256 x 16B chunks, 2 per thread
  int ar[2], ac[2];
#pragma unroll
  for (int i = 0; i < 2; ++i) {
    int c = tid + i * 128;            // chunk 0..255
    ar[i] = c >> 2;                   // row 0..63
    ac[i] = (c & 3) << 3;             // bf16 col {0,8,16,24}
  }

  v8f acc[4] = {{}, {}, {}, {}};

  for (int kb = 0; kb < DIM; kb += 32) {
    // ---- A tile: async global -> LDS (b128 per lane)
#pragma unroll
    for (int i = 0; i < 2; ++i) {
      unsigned lds_off = (unsigned)(size_t)&sA[ar[i] * 32 + ac[i]];
      const __bf16* g = &aA[(size_t)(mBase + ar[i]) * DIM + kb + ac[i]];
      asm volatile("global_load_async_to_lds_b128 %0, %1, off"
                   :: "v"(lds_off), "v"(g) : "memory");
    }
    // ---- B tile: float4 load, convert, ds_store_b64
    f4 vb[4];
#pragma unroll
    for (int i = 0; i < 4; ++i)
      vb[i] = *(const f4*)&proj_w[(size_t)(nBase + rr[i]) * DIM + kb + cc[i]];
    if (kb + 32 < DIM)
      __builtin_prefetch(&proj_w[(size_t)(nBase + rr[0]) * DIM + kb + 32 + cc[0]], 0, 0);
#pragma unroll
    for (int i = 0; i < 4; ++i)
      *(v4bf*)&sB[rr[i] * 32 + cc[i]] = cvt4(vb[i]);

    asm volatile("s_wait_asynccnt 0" ::: "memory");
    __syncthreads();

    v16bf a = load_frag(sA + wave * 16 * 32, 32, 0, lane);
#pragma unroll
    for (int nt = 0; nt < 4; ++nt) {
      v16bf bf = load_frag(sB + nt * 16 * 32, 32, 0, lane);
      acc[nt] = wmma_bf16(a, bf, acc[nt]);
    }
    __syncthreads();
  }

  int hi = (lane & 16) ? 8 : 0;
  int nl = lane & 15;
#pragma unroll
  for (int nt = 0; nt < 4; ++nt) {
#pragma unroll
    for (int v = 0; v < 8; ++v) {
      int grow = mBase + wave * 16 + v + hi;
      int gc   = nBase + nt * 16 + nl;
      out[(size_t)grow * DIM + gc] = acc[nt][v] + proj_b[gc];
    }
  }
}

// ---------------------------------------------------------------------------
// Host launcher
// ---------------------------------------------------------------------------
extern "C" void kernel_launch(void* const* d_in, const int* in_sizes, int n_in,
                              void* d_out, int out_size, void* d_ws, size_t ws_size,
                              hipStream_t stream) {
  const float* x       = (const float*)d_in[0];
  const float* coords  = (const float*)d_in[1];
  const float* elev    = (const float*)d_in[2];
  const float* qkv_w   = (const float*)d_in[3];
  const float* qkv_b   = (const float*)d_in[4];
  const float* proj_w  = (const float*)d_in[5];
  const float* proj_b  = (const float*)d_in[6];
  const float* btable  = (const float*)d_in[7];
  const float* alpha   = (const float*)d_in[8];
  float* out = (float*)d_out;

  char* ws = (char*)d_ws;
  const size_t sz_head = (size_t)BATCH * NUM_HEADS * SEQ_N * HEAD_DIM * sizeof(__bf16); // 6 MB
  __bf16* qbuf  = (__bf16*)(ws);
  __bf16* kbuf  = (__bf16*)(ws + sz_head);
  __bf16* vtbuf = (__bf16*)(ws + 2 * sz_head);
  __bf16* aout  = (__bf16*)(ws + 3 * sz_head);
  unsigned short* idx = (unsigned short*)(ws + 4 * sz_head);     // 8 MB

  // 1) relative-position bucket table
  {
    size_t total = (size_t)BATCH * SEQ_N * SEQ_N;
    int blocks = (int)((total + 255) / 256);
    bucket_idx_kernel<<<blocks, 256, 0, stream>>>(coords, idx);
  }
  // 2) QKV projection (writes q scaled, k, v^T in bf16)
  {
    dim3 grid(3 * DIM / 64, (BATCH * SEQ_N) / 64);   // 36 x 64
    qkv_gemm_kernel<<<grid, 128, 0, stream>>>(x, qkv_w, qkv_b, qbuf, kbuf, vtbuf);
  }
  // 3) flash attention with fused biases
  {
    int blocks = BATCH * NUM_HEADS * (SEQ_N / 64);   // 768
    attn_kernel<<<blocks, 128, 0, stream>>>(qbuf, kbuf, vtbuf, idx, btable,
                                            elev, alpha, aout);
  }
  // 4) output projection
  {
    dim3 grid(DIM / 64, (BATCH * SEQ_N) / 64);       // 12 x 64
    proj_gemm_kernel<<<grid, 128, 0, stream>>>(aout, proj_w, proj_b, out);
  }
}
